// MultiheadAttention_52201032515854
// MI455X (gfx1250) — matrix operand order repaired and checked by reading, not compiled, not run on hardware
//
#include <hip/hip_runtime.h>

#define Bn 2
#define Sn 2048
#define Hn 1024
#define NHn 16
#define HDn 64
#define Mn (Bn * Sn)   /* 4096 rows */
#define N3n (3 * Hn)   /* 3072 */

// 1/sqrt(HD) * log2(e): softmax computed in exp2 domain
#define QSCALE_LOG2E 0.18033688011112042f

typedef __attribute__((ext_vector_type(16))) __bf16 v16bf;
typedef __attribute__((ext_vector_type(8)))  __bf16 v8bf;
typedef __attribute__((ext_vector_type(8)))  float  v8f;

// float -> bf16 bits, round-to-nearest-even
static __device__ __forceinline__ unsigned short f2bf(float f) {
  unsigned int u = __builtin_bit_cast(unsigned int, f);
  u += 0x7FFFu + ((u >> 16) & 1u);
  return (unsigned short)(u >> 16);
}

// 16-byte vector load of 8 bf16 values
static __device__ __forceinline__ v8bf ld8(const unsigned short* p) {
  uint4 t = *reinterpret_cast<const uint4*>(p);
  return __builtin_bit_cast(v8bf, t);
}

static __device__ __forceinline__ v16bf cat8(v8bf lo, v8bf hi) {
  return __builtin_shufflevector(lo, hi, 0, 1, 2, 3, 4, 5, 6, 7, 8, 9, 10, 11, 12, 13, 14, 15);
}

// ---------------------------------------------------------------- convert (x4 vectorized)
__global__ void cvt_bf16_kernel(const float* __restrict__ in,
                                unsigned short* __restrict__ out, int n4) {
  int i = blockIdx.x * blockDim.x + threadIdx.x;
  if (i < n4) {
    const float4 f = reinterpret_cast<const float4*>(in)[i];
    ushort4 o;
    o.x = f2bf(f.x); o.y = f2bf(f.y); o.z = f2bf(f.z); o.w = f2bf(f.w);
    reinterpret_cast<ushort4*>(out)[i] = o;
  }
}

// ---------------------------------------------------------------- QKV GEMM (64x64/wave, 4x4 WMMA tiles)
// qkv[m][n] = sum_k x[m][k] * w_qkv[n][k] + b_qkv[n]
// Writes q (pre-scaled by log2e/sqrt(HD)), k as [B,NH,S,HD]; v transposed [B,NH,HD,S].
__global__ void __launch_bounds__(128) qkv_gemm_kernel(
    const unsigned short* __restrict__ xb, const unsigned short* __restrict__ wb,
    const float* __restrict__ bias, unsigned short* __restrict__ qb,
    unsigned short* __restrict__ kb, unsigned short* __restrict__ vtb) {
  const int NT = N3n / 64;  // 48 n-supertiles
  const int wave = threadIdx.x >> 5, lane = threadIdx.x & 31;
  const int tile = blockIdx.x * 4 + wave;
  const int nT = tile % NT, mT = tile / NT;
  const int m0 = mT * 64, n0 = nT * 64;
  const int mr = lane & 15, kh = lane >> 4;
  const unsigned short* arow0 = xb + (size_t)(m0 + mr) * Hn;
  const unsigned short* brow0 = wb + (size_t)(n0 + mr) * Hn;
  v8f acc[4][4] = {};
  for (int k0 = 0; k0 < Hn; k0 += 32) {
    v16bf a[4], b[4];
#pragma unroll
    for (int i = 0; i < 4; ++i) {
      const unsigned short* ar = arow0 + (size_t)(i * 16) * Hn;
      a[i] = cat8(ld8(ar + k0 + kh * 16), ld8(ar + k0 + kh * 16 + 8));
      const unsigned short* br = brow0 + (size_t)(i * 16) * Hn;
      b[i] = cat8(ld8(br + k0 + kh * 16), ld8(br + k0 + kh * 16 + 8));
    }
#pragma unroll
    for (int i = 0; i < 4; ++i)
#pragma unroll
      for (int j = 0; j < 4; ++j)
        acc[i][j] = __builtin_amdgcn_wmma_f32_16x16x32_bf16(
            false, a[i], false, b[j], (short)0, acc[i][j], false, false);
  }
#pragma unroll
  for (int j = 0; j < 4; ++j) {
    const int n = n0 + j * 16 + mr;  // D layout: col = lane&15
    const float bv = bias[n];
#pragma unroll
    for (int i = 0; i < 4; ++i) {
#pragma unroll
      for (int r = 0; r < 8; ++r) {
        const int m = m0 + i * 16 + r + kh * 8;  // lanes 0-15 rows 0-7, 16-31 rows 8-15
        const int bidx = m >> 11;                // / Sn
        const int s = m & (Sn - 1);
        const float val = acc[i][j][r] + bv;
        if (n < Hn) {  // Q, pre-scaled for exp2-domain softmax
          const int h = n >> 6, d = n & 63;
          qb[((size_t)(bidx * NHn + h) * Sn + s) * HDn + d] = f2bf(val * QSCALE_LOG2E);
        } else if (n < 2 * Hn) {  // K
          const int o = n - Hn, h = o >> 6, d = o & 63;
          kb[((size_t)(bidx * NHn + h) * Sn + s) * HDn + d] = f2bf(val);
        } else {  // V transposed per head: [HD][S]
          const int o = n - 2 * Hn, h = o >> 6, d = o & 63;
          vtb[((size_t)(bidx * NHn + h) * HDn + d) * Sn + s] = f2bf(val);
        }
      }
    }
  }
}

// ---------------------------------------------------------------- flash attention
// One 16-row q tile per wave; stream 32 keys/iteration with online softmax (exp2 domain).
__global__ void __launch_bounds__(128) flash_attn_kernel(
    const unsigned short* __restrict__ qb, const unsigned short* __restrict__ kb,
    const unsigned short* __restrict__ vtb, const float* __restrict__ mask,
    unsigned short* __restrict__ avb) {
  __shared__ alignas(16) unsigned short plds[4][16][32];  // per-wave P staging
  const int wave = threadIdx.x >> 5, lane = threadIdx.x & 31;
  const int tile = blockIdx.x * 4 + wave;
  const int qi = tile & (Sn / 16 - 1);  // q tile within head
  const int bh = tile >> 7;             // (b*NH + h)
  const unsigned short* qp = qb + (size_t)bh * Sn * HDn;
  const unsigned short* kp = kb + (size_t)bh * Sn * HDn;
  const unsigned short* vp = vtb + (size_t)bh * HDn * Sn;
  const int mr = lane & 15, kh = lane >> 4;

  // Q as WMMA A operand for the two d-steps (d=0..31, d=32..63), loaded once.
  const unsigned short* qrow = qp + (size_t)(qi * 16 + mr) * HDn;
  const v16bf aq0 = cat8(ld8(qrow + kh * 16), ld8(qrow + kh * 16 + 8));
  const v16bf aq1 = cat8(ld8(qrow + 32 + kh * 16), ld8(qrow + 32 + kh * 16 + 8));

  float mrow[8], lrow[8];
#pragma unroll
  for (int r = 0; r < 8; ++r) { mrow[r] = -1e30f; lrow[r] = 0.0f; }
  v8f o0 = {}, o1 = {}, o2 = {}, o3 = {};

  for (int k0 = 0; k0 < Sn; k0 += 32) {
    // ---- scores: two 16-key column tiles
    const unsigned short* kr0 = kp + (size_t)(k0 + mr) * HDn;
    const unsigned short* kr1 = kp + (size_t)(k0 + 16 + mr) * HDn;
    v16bf bk0lo = cat8(ld8(kr0 + kh * 16), ld8(kr0 + kh * 16 + 8));
    v16bf bk0hi = cat8(ld8(kr0 + 32 + kh * 16), ld8(kr0 + 32 + kh * 16 + 8));
    v16bf bk1lo = cat8(ld8(kr1 + kh * 16), ld8(kr1 + kh * 16 + 8));
    v16bf bk1hi = cat8(ld8(kr1 + 32 + kh * 16), ld8(kr1 + 32 + kh * 16 + 8));
    v8f s0 = {}, s1 = {};
    s0 = __builtin_amdgcn_wmma_f32_16x16x32_bf16(false, aq0, false, bk0lo, (short)0, s0, false, false);
    s0 = __builtin_amdgcn_wmma_f32_16x16x32_bf16(false, aq1, false, bk0hi, (short)0, s0, false, false);
    s1 = __builtin_amdgcn_wmma_f32_16x16x32_bf16(false, aq0, false, bk1lo, (short)0, s1, false, false);
    s1 = __builtin_amdgcn_wmma_f32_16x16x32_bf16(false, aq1, false, bk1hi, (short)0, s1, false, false);

    // ---- online softmax in exp2 domain (q carries log2e/sqrt(HD); mask gets it here)
    const float mk0 = mask[k0 + mr] * QSCALE_LOG2E;
    const float mk1 = mask[k0 + 16 + mr] * QSCALE_LOG2E;
#pragma unroll
    for (int r = 0; r < 8; ++r) {
      float a = s0[r] + mk0;
      float b = s1[r] + mk1;
      float mx = fmaxf(a, b);
      // row reduce across the 16 lanes of this half-wave (xor 1,2,4,8 stays in-half)
      mx = fmaxf(mx, __shfl_xor(mx, 1, 32));
      mx = fmaxf(mx, __shfl_xor(mx, 2, 32));
      mx = fmaxf(mx, __shfl_xor(mx, 4, 32));
      mx = fmaxf(mx, __shfl_xor(mx, 8, 32));
      const float mn = fmaxf(mrow[r], mx);
      const float al = exp2f(mrow[r] - mn);
      const float p0 = exp2f(a - mn);
      const float p1 = exp2f(b - mn);
      float rs = p0 + p1;
      rs += __shfl_xor(rs, 1, 32);
      rs += __shfl_xor(rs, 2, 32);
      rs += __shfl_xor(rs, 4, 32);
      rs += __shfl_xor(rs, 8, 32);
      lrow[r] = lrow[r] * al + rs;
      mrow[r] = mn;
      o0[r] *= al; o1[r] *= al; o2[r] *= al; o3[r] *= al;
      // stage P (D layout -> LDS, row-major 16x32)
      const int m = r + kh * 8;
      plds[wave][m][mr] = f2bf(p0);
      plds[wave][m][16 + mr] = f2bf(p1);
    }
    asm volatile("s_wait_dscnt 0" ::: "memory");  // LDS stores visible before reload

    // reload P in A-operand layout
    v16bf ap = cat8(ld8(&plds[wave][mr][kh * 16]), ld8(&plds[wave][mr][kh * 16 + 8]));

    // ---- PV: V^T rows are contiguous along k
    const unsigned short* vbase = vp + (size_t)mr * Sn + k0 + kh * 16;
    v16bf bv0 = cat8(ld8(vbase), ld8(vbase + 8));
    o0 = __builtin_amdgcn_wmma_f32_16x16x32_bf16(false, ap, false, bv0, (short)0, o0, false, false);
    v16bf bv1 = cat8(ld8(vbase + (size_t)16 * Sn), ld8(vbase + (size_t)16 * Sn + 8));
    o1 = __builtin_amdgcn_wmma_f32_16x16x32_bf16(false, ap, false, bv1, (short)0, o1, false, false);
    v16bf bv2 = cat8(ld8(vbase + (size_t)32 * Sn), ld8(vbase + (size_t)32 * Sn + 8));
    o2 = __builtin_amdgcn_wmma_f32_16x16x32_bf16(false, ap, false, bv2, (short)0, o2, false, false);
    v16bf bv3 = cat8(ld8(vbase + (size_t)48 * Sn), ld8(vbase + (size_t)48 * Sn + 8));
    o3 = __builtin_amdgcn_wmma_f32_16x16x32_bf16(false, ap, false, bv3, (short)0, o3, false, false);
  }

  // ---- epilogue: normalize and write av as bf16 [B,S,H]
  const int bb = bh / NHn, h = bh % NHn;
#pragma unroll
  for (int r = 0; r < 8; ++r) {
    const float inv = 1.0f / lrow[r];
    const int s = qi * 16 + r + kh * 8;
    unsigned short* orow = avb + ((size_t)bb * Sn + s) * Hn + h * HDn;
    orow[mr] = f2bf(o0[r] * inv);
    orow[16 + mr] = f2bf(o1[r] * inv);
    orow[32 + mr] = f2bf(o2[r] * inv);
    orow[48 + mr] = f2bf(o3[r] * inv);
  }
}

// ---------------------------------------------------------------- output GEMM (64x64/wave)
__global__ void __launch_bounds__(128) out_gemm_kernel(
    const unsigned short* __restrict__ ab, const unsigned short* __restrict__ wb,
    const float* __restrict__ bias, float* __restrict__ out) {
  const int NT = Hn / 64;  // 16 n-supertiles
  const int wave = threadIdx.x >> 5, lane = threadIdx.x & 31;
  const int tile = blockIdx.x * 4 + wave;
  const int nT = tile % NT, mT = tile / NT;
  const int m0 = mT * 64, n0 = nT * 64;
  const int mr = lane & 15, kh = lane >> 4;
  const unsigned short* arow0 = ab + (size_t)(m0 + mr) * Hn;
  const unsigned short* brow0 = wb + (size_t)(n0 + mr) * Hn;
  v8f acc[4][4] = {};
  for (int k0 = 0; k0 < Hn; k0 += 32) {
    v16bf a[4], b[4];
#pragma unroll
    for (int i = 0; i < 4; ++i) {
      const unsigned short* ar = arow0 + (size_t)(i * 16) * Hn;
      a[i] = cat8(ld8(ar + k0 + kh * 16), ld8(ar + k0 + kh * 16 + 8));
      const unsigned short* br = brow0 + (size_t)(i * 16) * Hn;
      b[i] = cat8(ld8(br + k0 + kh * 16), ld8(br + k0 + kh * 16 + 8));
    }
#pragma unroll
    for (int i = 0; i < 4; ++i)
#pragma unroll
      for (int j = 0; j < 4; ++j)
        acc[i][j] = __builtin_amdgcn_wmma_f32_16x16x32_bf16(
            false, a[i], false, b[j], (short)0, acc[i][j], false, false);
  }
#pragma unroll
  for (int j = 0; j < 4; ++j) {
    const int n = n0 + j * 16 + mr;
    const float bv = bias[n];
#pragma unroll
    for (int i = 0; i < 4; ++i) {
#pragma unroll
      for (int r = 0; r < 8; ++r) {
        const int m = m0 + i * 16 + r + kh * 8;
        out[(size_t)m * Hn + n] = acc[i][j][r] + bv;
      }
    }
  }
}

// ---------------------------------------------------------------- launch
extern "C" void kernel_launch(void* const* d_in, const int* in_sizes, int n_in,
                              void* d_out, int out_size, void* d_ws, size_t ws_size,
                              hipStream_t stream) {
  const float* x = (const float*)d_in[0];
  const float* mask = (const float*)d_in[1];
  const float* w_qkv = (const float*)d_in[2];
  const float* b_qkv = (const float*)d_in[3];
  const float* w_o = (const float*)d_in[4];
  const float* b_o = (const float*)d_in[5];
  float* out = (float*)d_out;

  char* ws = (char*)d_ws;  // 48 MB used
  unsigned short* xb    = (unsigned short*)(ws);
  unsigned short* wqkvb = (unsigned short*)(ws + (size_t)8 * 1024 * 1024);
  unsigned short* wob   = (unsigned short*)(ws + (size_t)14 * 1024 * 1024);
  unsigned short* qbuf  = (unsigned short*)(ws + (size_t)16 * 1024 * 1024);
  unsigned short* kbuf  = (unsigned short*)(ws + (size_t)24 * 1024 * 1024);
  unsigned short* vtbuf = (unsigned short*)(ws + (size_t)32 * 1024 * 1024);
  unsigned short* avbuf = (unsigned short*)(ws + (size_t)40 * 1024 * 1024);

  cvt_bf16_kernel<<<(Mn * Hn / 4 + 255) / 256, 256, 0, stream>>>(x, xb, Mn * Hn / 4);
  cvt_bf16_kernel<<<(N3n * Hn / 4 + 255) / 256, 256, 0, stream>>>(w_qkv, wqkvb, N3n * Hn / 4);
  cvt_bf16_kernel<<<(Hn * Hn / 4 + 255) / 256, 256, 0, stream>>>(w_o, wob, Hn * Hn / 4);

  qkv_gemm_kernel<<<(Mn / 64) * (N3n / 64) / 4, 128, 0, stream>>>(xb, wqkvb, b_qkv, qbuf, kbuf, vtbuf);
  flash_attn_kernel<<<(Bn * NHn * (Sn / 16)) / 4, 128, 0, stream>>>(qbuf, kbuf, vtbuf, mask, avbuf);
  out_gemm_kernel<<<(Mn / 64) * (Hn / 64) / 4, 128, 0, stream>>>(avbuf, wob, b_o, out);
}
